// Zeoformer_72962904425134
// MI455X (gfx1250) — matrix operand order repaired and checked by reading, not compiled, not run on hardware
//
#include <hip/hip_runtime.h>

#define N_NODES 20000
#define N_EDGES 320000
#define NBATCH  64
#define FDIM    256
#define CGDIM   92
#define NPPE    3

typedef __attribute__((ext_vector_type(16))) __bf16 v16bf;
typedef __attribute__((ext_vector_type(8)))  __bf16 v8bf;
typedef __attribute__((ext_vector_type(8)))  float  v8f;

__device__ __forceinline__ float act_apply(float x, int act) {
  if (act == 1) return x / (1.0f + expf(-x));                              // SiLU
  if (act == 2) return 0.5f * x * (1.0f + erff(x * 0.7071067811865476f));  // GELU exact
  return x;
}

// ---------------------------------------------------------------------------
// Fragment loaders (CDNA5 ISA 7.12.2 wave32 layouts).
// A 16x32 bf16 fragment for one lane = two runs of 8 consecutive k values:
//   a[0..7]  = A[row, kb + hh*8 .. +7]
//   a[8..15] = A[row, kb + 16 + hh*8 .. +7]
// B 32x16 fragment = same two runs of 8 k values of column n:
//   b[0..7] = W[n, kb + hh*8 .. +7], b[8..15] = W[n, kb + 16 + hh*8 .. +7]
// ---------------------------------------------------------------------------
__device__ __forceinline__ v16bf load_a_frag(const float* __restrict__ A,
                                             long long rowbase, int koff) {
  const float* p = A + rowbase + koff;
  float4 r0 = *(const float4*)(p);
  float4 r1 = *(const float4*)(p + 4);
  float4 r2 = *(const float4*)(p + 16);
  float4 r3 = *(const float4*)(p + 20);
  v16bf a;
  a[0]=(__bf16)r0.x;  a[1]=(__bf16)r0.y;  a[2]=(__bf16)r0.z;  a[3]=(__bf16)r0.w;
  a[4]=(__bf16)r1.x;  a[5]=(__bf16)r1.y;  a[6]=(__bf16)r1.z;  a[7]=(__bf16)r1.w;
  a[8]=(__bf16)r2.x;  a[9]=(__bf16)r2.y;  a[10]=(__bf16)r2.z; a[11]=(__bf16)r2.w;
  a[12]=(__bf16)r3.x; a[13]=(__bf16)r3.y; a[14]=(__bf16)r3.z; a[15]=(__bf16)r3.w;
  return a;
}

__device__ __forceinline__ v16bf load_a_frag(const __bf16* __restrict__ A,
                                             long long rowbase, int koff) {
  const __bf16* p = A + rowbase + koff;
  v8bf r0 = *(const v8bf*)(p);        // 16B aligned: lda=256, koff mult of 8
  v8bf r1 = *(const v8bf*)(p + 16);
  v16bf a;
#pragma unroll
  for (int i = 0; i < 8; ++i) { a[i] = r0[i]; a[8 + i] = r1[i]; }
  return a;
}

__device__ __forceinline__ v16bf load_b_frag(const float* __restrict__ W,
                                             long long rowbase, int koff) {
  const float* p = W + rowbase + koff;
  float4 q0 = *(const float4*)(p);
  float4 q1 = *(const float4*)(p + 4);
  float4 q2 = *(const float4*)(p + 16);
  float4 q3 = *(const float4*)(p + 20);
  v16bf b;
  b[0]=(__bf16)q0.x;  b[1]=(__bf16)q0.y;  b[2]=(__bf16)q0.z;  b[3]=(__bf16)q0.w;
  b[4]=(__bf16)q1.x;  b[5]=(__bf16)q1.y;  b[6]=(__bf16)q1.z;  b[7]=(__bf16)q1.w;
  b[8]=(__bf16)q2.x;  b[9]=(__bf16)q2.y;  b[10]=(__bf16)q2.z; b[11]=(__bf16)q2.w;
  b[12]=(__bf16)q3.x; b[13]=(__bf16)q3.y; b[14]=(__bf16)q3.z; b[15]=(__bf16)q3.w;
  return b;
}

template <typename AT>
__device__ __forceinline__ v16bf load_a_tail(const AT* __restrict__ A,
                                             long long rowbase, int kb, int hh,
                                             int K) {
  v16bf a;
#pragma unroll
  for (int i = 0; i < 8; ++i) {
    int k0 = kb + hh * 8 + i;
    int k1 = kb + 16 + hh * 8 + i;
    a[i]     = (k0 < K) ? (__bf16)(float)A[rowbase + k0] : (__bf16)0.0f;
    a[8 + i] = (k1 < K) ? (__bf16)(float)A[rowbase + k1] : (__bf16)0.0f;
  }
  return a;
}

__device__ __forceinline__ v16bf load_b_tail(const float* __restrict__ W,
                                             long long rowbase, int kb, int hh,
                                             int K) {
  v16bf b;
#pragma unroll
  for (int i = 0; i < 16; ++i) {
    int k = kb + (i < 8 ? hh * 8 + i : 16 + hh * 8 + (i - 8));
    b[i] = (k < K) ? (__bf16)W[rowbase + k] : (__bf16)0.0f;
  }
  return b;
}

// ---------------------------------------------------------------------------
// WMMA GEMM: C[M,Nout] = act(A[M,K] @ W^T + bias). W row-major [Nout,K].
// REQUIRES M % 16 == 0 and Nout % 64 == 0 (true for all call sites).
// One wave computes a 16x64 strip: 1 A-fragment feeds 4 WMMAs per K-step.
// ---------------------------------------------------------------------------
template <typename AT, typename OT>
__global__ __launch_bounds__(256)
void gemm_wmma_kernel(const AT* __restrict__ A, int lda,
                      const float* __restrict__ W,
                      const float* __restrict__ bias,
                      OT* __restrict__ C, int ldc,
                      int M, int K, int Nout, int act) {
  const int lane = threadIdx.x & 31;
  const int wid  = threadIdx.x >> 5;
  const int NG = Nout >> 6;                 // 16x64 strips along N
  const int T  = (M >> 4) * NG;
  const int t  = blockIdx.x * (blockDim.x >> 5) + wid;
  if (t >= T) return;                       // wave-uniform
  const int mtile = t / NG;
  const int ng    = t % NG;
  const int hh  = lane >> 4;
  const int l16 = lane & 15;

  const long long arow = (long long)(mtile * 16 + l16) * lda;
  const long long wr0 = (long long)(ng * 64 + l16) * K;
  const long long wr1 = wr0 + (long long)16 * K;
  const long long wr2 = wr0 + (long long)32 * K;
  const long long wr3 = wr0 + (long long)48 * K;

  v8f c0 = {}, c1 = {}, c2 = {}, c3 = {};
  const int Kfull = K & ~31;
  for (int kb = 0; kb < Kfull; kb += 32) {
    const int ka = kb + hh * 8;
    const int kw = kb + hh * 8;
    v16bf a = load_a_frag(A, arow, ka);
    v16bf b0 = load_b_frag(W, wr0, kw);
    v16bf b1 = load_b_frag(W, wr1, kw);
    v16bf b2 = load_b_frag(W, wr2, kw);
    v16bf b3 = load_b_frag(W, wr3, kw);
    c0 = __builtin_amdgcn_wmma_f32_16x16x32_bf16(false, a, false, b0, (short)0, c0, false, false);
    c1 = __builtin_amdgcn_wmma_f32_16x16x32_bf16(false, a, false, b1, (short)0, c1, false, false);
    c2 = __builtin_amdgcn_wmma_f32_16x16x32_bf16(false, a, false, b2, (short)0, c2, false, false);
    c3 = __builtin_amdgcn_wmma_f32_16x16x32_bf16(false, a, false, b3, (short)0, c3, false, false);
  }
  if (Kfull < K) {                          // ragged tail (K=92 cases only)
    v16bf a = load_a_tail(A, arow, Kfull, hh, K);
    v16bf b0 = load_b_tail(W, wr0, Kfull, hh, K);
    v16bf b1 = load_b_tail(W, wr1, Kfull, hh, K);
    v16bf b2 = load_b_tail(W, wr2, Kfull, hh, K);
    v16bf b3 = load_b_tail(W, wr3, Kfull, hh, K);
    c0 = __builtin_amdgcn_wmma_f32_16x16x32_bf16(false, a, false, b0, (short)0, c0, false, false);
    c1 = __builtin_amdgcn_wmma_f32_16x16x32_bf16(false, a, false, b1, (short)0, c1, false, false);
    c2 = __builtin_amdgcn_wmma_f32_16x16x32_bf16(false, a, false, b2, (short)0, c2, false, false);
    c3 = __builtin_amdgcn_wmma_f32_16x16x32_bf16(false, a, false, b3, (short)0, c3, false, false);
  }

  // C layout: VGPR r -> row = mtile*16 + r + hh*8, col = n
  const int rowb = mtile * 16 + hh * 8;
  const int col0 = ng * 64 + l16;
#pragma unroll
  for (int r = 0; r < 8; ++r) {
    const long long ro = (long long)(rowb + r) * ldc;
    float v0 = act_apply(c0[r] + (bias ? bias[col0]      : 0.0f), act);
    float v1 = act_apply(c1[r] + (bias ? bias[col0 + 16] : 0.0f), act);
    float v2 = act_apply(c2[r] + (bias ? bias[col0 + 32] : 0.0f), act);
    float v3 = act_apply(c3[r] + (bias ? bias[col0 + 48] : 0.0f), act);
    C[ro + col0]      = (OT)v0;
    C[ro + col0 + 16] = (OT)v1;
    C[ro + col0 + 32] = (OT)v2;
    C[ro + col0 + 48] = (OT)v3;
  }
}

// ---------------------------------------------------------------------------
// Elementwise / graph kernels
// ---------------------------------------------------------------------------
__global__ void preproc_kernel(const float* __restrict__ x, float* __restrict__ xf,
                               float* __restrict__ xp, int n) {
  int i = blockIdx.x * blockDim.x + threadIdx.x;
  if (i >= n) return;
  float v = x[i];
  xf[i] = floorf(v);
  xp[i] = (v - truncf(v)) * 1000.0f;
}

__global__ void fill_kernel(float* __restrict__ p, float v, int n) {
  int i = blockIdx.x * blockDim.x + threadIdx.x;
  if (i < n) p[i] = v;
}

__global__ void add2_kernel(float* __restrict__ out, const float* __restrict__ a,
                            const float* __restrict__ b, int n) {
  int i = blockIdx.x * blockDim.x + threadIdx.x;
  if (i < n) out[i] = a[i] + b[i];
}

__global__ void rbf_kernel(const float* __restrict__ attr, float* __restrict__ rbf,
                           int E) {
  int eid = blockIdx.x;
  int f = threadIdx.x;
  if (eid >= E) return;
  float a0 = attr[eid * 3 + 0], a1 = attr[eid * 3 + 1], a2 = attr[eid * 3 + 2];
  float d = -1.0f / sqrtf(a0 * a0 + a1 * a1 + a2 * a2);
  float cen = -6.0f + 6.0f * (float)f / (float)(FDIM - 1);
  const float gamma = (float)(FDIM - 1) / 6.0f;  // 42.5
  float t = d - cen;
  rbf[(long long)eid * FDIM + f] = expf(-gamma * t * t);
}

// s[e] = dot(q[dst], k[src]+e_feat[e]) / sqrt(F); atomic CAS-max into mseg[dst]
__global__ __launch_bounds__(256)
void edge_score_kernel(const float* __restrict__ q, const float* __restrict__ k,
                       const float* __restrict__ ef, const int* __restrict__ src,
                       const int* __restrict__ dst, float* __restrict__ s,
                       float* __restrict__ mseg, int E) {
  int lane = threadIdx.x & 31;
  int wid  = threadIdx.x >> 5;
  int eid = blockIdx.x * (blockDim.x >> 5) + wid;
  if (eid >= E) return;
  int sd = src[eid], dd = dst[eid];
  const float* qr = q + (long long)dd * FDIM + lane * 8;
  const float* kr = k + (long long)sd * FDIM + lane * 8;
  const float* er = ef + (long long)eid * FDIM + lane * 8;
  float4 q0 = *(const float4*)(qr),     q1 = *(const float4*)(qr + 4);
  float4 k0 = *(const float4*)(kr),     k1 = *(const float4*)(kr + 4);
  float4 e0 = *(const float4*)(er),     e1 = *(const float4*)(er + 4);
  float acc = q0.x * (k0.x + e0.x) + q0.y * (k0.y + e0.y) +
              q0.z * (k0.z + e0.z) + q0.w * (k0.w + e0.w) +
              q1.x * (k1.x + e1.x) + q1.y * (k1.y + e1.y) +
              q1.z * (k1.z + e1.z) + q1.w * (k1.w + e1.w);
  for (int off = 16; off; off >>= 1) acc += __shfl_xor(acc, off, 32);
  if (lane == 0) {
    float sv = acc * (1.0f / 16.0f);
    s[eid] = sv;
    float old = mseg[dd];
    while (sv > old) {
      int prev = atomicCAS((int*)(mseg + dd), __float_as_int(old),
                           __float_as_int(sv));
      float pf = __int_as_float(prev);
      if (pf == old) break;
      old = pf;
    }
  }
}

__global__ void edge_exp_kernel(float* __restrict__ s, const float* __restrict__ mseg,
                                float* __restrict__ den, const int* __restrict__ dst,
                                int E) {
  int e = blockIdx.x * blockDim.x + threadIdx.x;
  if (e >= E) return;
  int dd = dst[e];
  float ex = expf(s[e] - mseg[dd]);
  s[e] = ex;
  atomicAdd(den + dd, ex);
}

// out[dst] += (s[e]/den[dst]) * (v[src] + e_feat[e]); one block per edge
__global__ __launch_bounds__(256)
void edge_agg_kernel(const float* __restrict__ s, const float* __restrict__ den,
                     const float* __restrict__ v, const float* __restrict__ ef,
                     const int* __restrict__ src, const int* __restrict__ dst,
                     float* __restrict__ out, int E) {
  int eid = blockIdx.x;
  int f = threadIdx.x;
  if (eid >= E) return;
  int sd = src[eid], dd = dst[eid];
  float alpha = s[eid] / den[dd];
  float val = alpha * (v[(long long)sd * FDIM + f] + ef[(long long)eid * FDIM + f]);
  atomicAdd(out + (long long)dd * FDIM + f, val);
}

__global__ void bn_kernel(float* __restrict__ out, const float* __restrict__ adj,
                          const float* __restrict__ g, const float* __restrict__ beta,
                          int total) {
  int i = blockIdx.x * blockDim.x + threadIdx.x;
  if (i >= total) return;
  int f = i & (FDIM - 1);
  out[i] = adj[i] * g[f] + beta[f];
}

// x1 *= gelu(a2[:, F:])   (a2 row stride 2F)
__global__ void gprod_kernel(float* __restrict__ x1, const float* __restrict__ a2,
                             int total) {
  int i = blockIdx.x * blockDim.x + threadIdx.x;
  if (i >= total) return;
  int row = i >> 8;
  int f = i & (FDIM - 1);
  float x2 = a2[(long long)row * (2 * FDIM) + FDIM + f];
  float g = 0.5f * x2 * (1.0f + erff(x2 * 0.7071067811865476f));
  x1[i] *= g;
}

__global__ void count_kernel(const int* __restrict__ batch, float* __restrict__ cnt,
                             int n) {
  int i = blockIdx.x * blockDim.x + threadIdx.x;
  if (i < n) atomicAdd(cnt + batch[i], 1.0f);
}

__global__ void pool_kernel(const float* __restrict__ h, const int* __restrict__ batch,
                            float* __restrict__ tot, int total) {
  int i = blockIdx.x * blockDim.x + threadIdx.x;
  if (i >= total) return;
  int node = i >> 8;
  int f = i & (FDIM - 1);
  atomicAdd(tot + (long long)batch[node] * FDIM + f, h[i]);
}

__global__ void divcnt_kernel(float* __restrict__ tot, const float* __restrict__ cnt,
                              int total) {
  int i = blockIdx.x * blockDim.x + threadIdx.x;
  if (i >= total) return;
  int b = i >> 8;
  tot[i] /= fmaxf(cnt[b], 1.0f);
}

__global__ void final_kernel(const float* __restrict__ crystal,
                             const float* __restrict__ outW,
                             const float* __restrict__ outb,
                             const float* __restrict__ nosda,
                             float* __restrict__ out) {
  int b = threadIdx.x;
  if (b >= NBATCH) return;
  float acc = 0.f;
  for (int f = 0; f < FDIM; ++f) acc += crystal[b * FDIM + f] * outW[f];
  out[b] = (acc + outb[0]) / nosda[b];
}

// ---------------------------------------------------------------------------
// Host orchestration
// ---------------------------------------------------------------------------
static inline int cdiv(int a, int b) { return (a + b - 1) / b; }

extern "C" void kernel_launch(void* const* d_in, const int* in_sizes, int n_in,
                              void* d_out, int out_size, void* d_ws, size_t ws_size,
                              hipStream_t stream) {
  (void)in_sizes; (void)n_in; (void)out_size; (void)ws_size;
  const int N = N_NODES, E = N_EDGES, F = FDIM, CG = CGDIM, B = NBATCH;

  const float* x     = (const float*)d_in[0];
  const int*   eidx  = (const int*)d_in[1];
  const float* eattr = (const float*)d_in[2];
  const int*   batch = (const int*)d_in[3];
  const float* nosda = (const float*)d_in[4];
  const float* ae_W1 = (const float*)d_in[5];
  const float* ae_b1 = (const float*)d_in[6];
  const float* ae_W2 = (const float*)d_in[7];
  const float* ae_b2 = (const float*)d_in[8];
  const float* ee_W  = (const float*)d_in[9];
  const float* ee_b  = (const float*)d_in[10];
  const float* pe_W  = (const float*)d_in[11];
  const float* pe_b  = (const float*)d_in[12];
  const float* Wq = (const float*)d_in[13]; const float* bq = (const float*)d_in[14];
  const float* Wk = (const float*)d_in[15]; const float* bk = (const float*)d_in[16];
  const float* Wv = (const float*)d_in[17]; const float* bv = (const float*)d_in[18];
  const float* We = (const float*)d_in[19]; const float* be = (const float*)d_in[20];
  const float* Ws = (const float*)d_in[21]; const float* bs = (const float*)d_in[22];
  const float* pW1 = (const float*)d_in[23]; const float* pb1 = (const float*)d_in[24];
  const float* pW2 = (const float*)d_in[25]; const float* pb2 = (const float*)d_in[26];
  const float* pW3 = (const float*)d_in[27]; const float* pb3 = (const float*)d_in[28];
  const float* pg    = (const float*)d_in[29];
  const float* pbeta = (const float*)d_in[30];
  const float* fc_W  = (const float*)d_in[31];
  const float* fc_b  = (const float*)d_in[32];
  const float* out_W = (const float*)d_in[33];
  const float* out_b = (const float*)d_in[34];

  const int* src = eidx;
  const int* dst = eidx + E;

  // ---- workspace bump allocator (~680 MB) ----
  char* wp = (char*)d_ws;
  auto alloc = [&](size_t bytes) -> void* {
    void* r = (void*)wp;
    wp += (bytes + 255) & ~(size_t)255;
    return r;
  };
  float* xf    = (float*)alloc((size_t)N * CG * 4);
  float* xp    = (float*)alloc((size_t)N * CG * 4);
  float* hbuf  = (float*)alloc((size_t)N * F * 4);
  float* hnew  = (float*)alloc((size_t)N * F * 4);
  float* ppe   = (float*)alloc((size_t)N * F * 4);
  float* qb    = (float*)alloc((size_t)N * F * 4);
  float* kb    = (float*)alloc((size_t)N * F * 4);
  float* vb    = (float*)alloc((size_t)N * F * 4);
  float* a2b   = (float*)alloc((size_t)N * 2 * F * 4);
  __bf16* efb  = (__bf16*)alloc((size_t)E * F * 2);
  float* ebig  = (float*)alloc((size_t)E * F * 4);   // rbf, then per-layer e
  float* sbuf  = (float*)alloc((size_t)E * 4);
  float* mseg  = (float*)alloc((size_t)N * 4);
  float* den   = (float*)alloc((size_t)N * 4);
  float* tot   = (float*)alloc((size_t)B * F * 4);
  float* cnt   = (float*)alloc((size_t)B * 4);
  float* cbuf  = (float*)alloc((size_t)B * F * 4);

  // GEMM launchers: 8 waves/block, each wave one 16x64 strip
  auto G_ff = [&](const float* A, int lda, const float* W, const float* bias,
                  float* C, int ldc, int M, int K, int Nout, int act) {
    int T = (M >> 4) * (Nout >> 6);
    gemm_wmma_kernel<float, float><<<cdiv(T, 8), 256, 0, stream>>>(
        A, lda, W, bias, C, ldc, M, K, Nout, act);
  };
  auto G_fb = [&](const float* A, int lda, const float* W, const float* bias,
                  __bf16* C, int ldc, int M, int K, int Nout, int act) {
    int T = (M >> 4) * (Nout >> 6);
    gemm_wmma_kernel<float, __bf16><<<cdiv(T, 8), 256, 0, stream>>>(
        A, lda, W, bias, C, ldc, M, K, Nout, act);
  };
  auto G_bf = [&](const __bf16* A, int lda, const float* W, const float* bias,
                  float* C, int ldc, int M, int K, int Nout, int act) {
    int T = (M >> 4) * (Nout >> 6);
    gemm_wmma_kernel<__bf16, float><<<cdiv(T, 8), 256, 0, stream>>>(
        A, lda, W, bias, C, ldc, M, K, Nout, act);
  };

  // ---- embeddings ----
  preproc_kernel<<<cdiv(N * CG, 256), 256, 0, stream>>>(x, xf, xp, N * CG);
  G_ff(xf, CG, ae_W1, ae_b1, qb, F, N, CG, F, 1);      // silu(floor(x)@W1^T+b1)
  G_ff(qb, F, ae_W2, ae_b2, hbuf, F, N, F, F, 0);      // h
  G_ff(xp, CG, pe_W, pe_b, ppe, F, N, CG, F, 0);       // ppe
  rbf_kernel<<<E, 256, 0, stream>>>(eattr, ebig, E);   // rbf -> ebig
  G_fb(ebig, F, ee_W, ee_b, efb, F, E, F, F, 1);       // ef = silu(rbf@W^T+b), bf16

  // ---- 4 transformer conv layers (ppe MLP after first 3) ----
  for (int i = 0; i < NPPE + 1; ++i) {
    const float* Wq_i = Wq + (size_t)i * F * F; const float* bq_i = bq + (size_t)i * F;
    const float* Wk_i = Wk + (size_t)i * F * F; const float* bk_i = bk + (size_t)i * F;
    const float* Wv_i = Wv + (size_t)i * F * F; const float* bv_i = bv + (size_t)i * F;
    const float* We_i = We + (size_t)i * F * F; const float* be_i = be + (size_t)i * F;
    const float* Ws_i = Ws + (size_t)i * F * F; const float* bs_i = bs + (size_t)i * F;

    G_ff(hbuf, F, Wq_i, bq_i, qb, F, N, F, F, 0);
    G_ff(hbuf, F, Wk_i, bk_i, kb, F, N, F, F, 0);
    G_ff(hbuf, F, Wv_i, bv_i, vb, F, N, F, F, 0);
    G_bf(efb, F, We_i, be_i, ebig, F, E, F, F, 0);     // e = ef@We^T+be
    G_ff(hbuf, F, Ws_i, bs_i, hnew, F, N, F, F, 0);    // skip term (agg target)

    fill_kernel<<<cdiv(N, 256), 256, 0, stream>>>(mseg, -3.0e38f, N);
    fill_kernel<<<cdiv(N, 256), 256, 0, stream>>>(den, 0.0f, N);
    edge_score_kernel<<<cdiv(E, 8), 256, 0, stream>>>(qb, kb, ebig, src, dst,
                                                      sbuf, mseg, E);
    edge_exp_kernel<<<cdiv(E, 256), 256, 0, stream>>>(sbuf, mseg, den, dst, E);
    edge_agg_kernel<<<E, 256, 0, stream>>>(sbuf, den, vb, ebig, src, dst, hnew, E);
    // hnew = conv(h) output

    if (i < NPPE) {
      const float* pW1_i = pW1 + (size_t)i * 2 * F * F;
      const float* pb1_i = pb1 + (size_t)i * 2 * F;
      const float* pW2_i = pW2 + (size_t)i * F * F;
      const float* pb2_i = pb2 + (size_t)i * F;
      const float* pW3_i = pW3 + (size_t)i * F * F;
      const float* pb3_i = pb3 + (size_t)i * F;

      add2_kernel<<<cdiv(N * F, 256), 256, 0, stream>>>(ppe, ppe, hnew, N * F);
      bn_kernel<<<cdiv(N * F, 256), 256, 0, stream>>>(qb, ppe, pg + (size_t)i * F,
                                                      pbeta + (size_t)i * F, N * F);
      G_ff(qb, F, pW1_i, pb1_i, a2b, 2 * F, N, F, 2 * F, 0);   // a2 [N,512]
      G_ff(a2b, 2 * F, pW2_i, pb2_i, kb, F, N, F, F, 0);       // x1
      gprod_kernel<<<cdiv(N * F, 256), 256, 0, stream>>>(kb, a2b, N * F);
      G_ff(kb, F, pW3_i, pb3_i, vb, F, N, F, F, 0);
      add2_kernel<<<cdiv(N * F, 256), 256, 0, stream>>>(hbuf, vb, hnew, N * F);
      // hbuf = next h
    }
  }
  // final h is in hnew

  // ---- scatter-mean pooling + head ----
  fill_kernel<<<cdiv(B * F, 256), 256, 0, stream>>>(tot, 0.0f, B * F);
  fill_kernel<<<1, 256, 0, stream>>>(cnt, 0.0f, B);
  count_kernel<<<cdiv(N, 256), 256, 0, stream>>>(batch, cnt, N);
  pool_kernel<<<cdiv(N * F, 256), 256, 0, stream>>>(hnew, batch, tot, N * F);
  divcnt_kernel<<<cdiv(B * F, 256), 256, 0, stream>>>(tot, cnt, B * F);
  G_ff(tot, F, fc_W, fc_b, cbuf, F, B, F, F, 1);              // silu(tot@fc^T+b)
  add2_kernel<<<cdiv(B * F, 256), 256, 0, stream>>>(cbuf, cbuf, tot, B * F);
  final_kernel<<<1, 64, 0, stream>>>(cbuf, out_W, out_b, nosda, (float*)d_out);
}
